// ConditionalFlow_46127948759642
// MI455X (gfx1250) — matrix operand order repaired and checked by reading, not compiled, not run on hardware
//
#include <hip/hip_runtime.h>

typedef _Float16 v16h __attribute__((ext_vector_type(16)));
typedef _Float16 v8h  __attribute__((ext_vector_type(8)));
typedef float    v8f  __attribute__((ext_vector_type(8)));
typedef float    v4f  __attribute__((ext_vector_type(4)));

#define ROWS_PER_WG 128
#define THREADS     256
#define HSTRIDE     72        // activation row stride in halves (bank-conflict pad)
#define OSTRIDE     132       // outb row stride in floats (bank-conflict pad)
#define LDS_BYTES   140288

// LDS layout offsets (bytes)
#define OFF_X    0            // 128*8 f32             4096
#define OFF_C    4096         // 128*4 f32             2048
#define OFF_W1   6144         // 4 grp *512 halves     4096  (B-order, K padded 8->32)
#define OFF_W2   10240        // 8 (grp,ks)*512 halves 8192  (B-order)
#define OFF_W3   18432        // 16 (grp,ks)*512 h    16384  (B-order, cols permuted)
#define OFF_B12  34816        // 128 f32                512
#define OFF_B3   35328        // 128 f32 (permuted)     512
#define OFF_HA   35840        // 128*72 f16           18432
#define OFF_HB   54272        // 128*72 f16           18432
#define OFF_OUT  72704        // 128*132 f32          67584
                              // total               140288

// Conditioner / transform dim tables derived from build_masks() in the reference.
__constant__ int COND_TBL[6][4] = {
  {1,3,5,7},{0,2,4,6},{2,3,6,7},{0,1,4,5},{4,5,6,7},{0,1,2,3}};
__constant__ int TRAFO_TBL[6][4] = {
  {0,2,4,6},{1,3,5,7},{0,1,4,5},{2,3,6,7},{0,1,2,3},{4,5,6,7}};

__device__ __forceinline__ v8f wmma_f16(v16h a, v16h b, v8f c) {
  return __builtin_amdgcn_wmma_f32_16x16x32_f16(
      false, a, false, b, (short)0, c, false, false);
}

__device__ __forceinline__ v16h cat8(v8h lo, v8h hi) {
  return __builtin_shufflevector(lo, hi, 0,1,2,3,4,5,6,7,8,9,10,11,12,13,14,15);
}

// leaky_relu(t, 0.01) == max(t, 0.01*t)
__device__ __forceinline__ float lrelu(float t) {
  return fmaxf(t, 0.01f * t);
}

// stable softplus with native exp/log only
__device__ __forceinline__ float softplusf(float x) {
  return fmaxf(x, 0.0f) + __logf(1.0f + __expf(-fabsf(x)));
}

// B-operand swizzle: element (k within 64-K block, col within 64-col panel)
// -> dst half index inside a [(grp,ks)][lane][16] region of 512-half pages.
__device__ __forceinline__ int bswiz(int k, int col) {
  const int page = (col >> 4) * 2 + (k >> 5);      // (colgroup, kstep)
  const int lane = (col & 15) + (k & 16);          // B layout: lanes 0-15 K0..15, 16-31 K16..31
  const int h    = k & 15;
  return page * 512 + lane * 16 + h;
}

__global__ void __launch_bounds__(THREADS, 1)
flow_wmma_kernel(const float* __restrict__ xg, const float* __restrict__ cg,
                 const float* __restrict__ W1g, const float* __restrict__ b1g,
                 const float* __restrict__ W2g, const float* __restrict__ b2g,
                 const float* __restrict__ W3g, const float* __restrict__ b3g,
                 float* __restrict__ jac_out)
{
  extern __shared__ char smem[];
  float*    lds_x = (float*)(smem + OFF_X);
  float*    lds_c = (float*)(smem + OFF_C);
  _Float16* w1s   = (_Float16*)(smem + OFF_W1);
  _Float16* w2s   = (_Float16*)(smem + OFF_W2);
  _Float16* w3s   = (_Float16*)(smem + OFF_W3);
  float*    b12   = (float*)(smem + OFF_B12);
  float*    b3s   = (float*)(smem + OFF_B3);
  _Float16* hA    = (_Float16*)(smem + OFF_HA);
  _Float16* hB    = (_Float16*)(smem + OFF_HB);
  float*    outb  = (float*)(smem + OFF_OUT);

  const int tid  = threadIdx.x;
  const int lane = tid & 31;
  const int wave = tid >> 5;
  const int row0 = blockIdx.x * ROWS_PER_WG;

  for (int idx = tid; idx < ROWS_PER_WG * 8; idx += THREADS)
    lds_x[idx] = xg[row0 * 8 + idx];
  for (int idx = tid; idx < ROWS_PER_WG * 4; idx += THREADS)
    lds_c[idx] = cg[row0 * 4 + idx];

  const int rA   = lane & 15;
  const int rWG  = wave * 16 + rA;
  const int mHi  = (lane & 16) ? 8 : 0;    // A K-offset / C-D row offset for hi lanes
  const int colL = lane & 15;

  const float DINV = 1.0f / (0.001f + 0.69314718055994531f); // 1/(MIN_D + ln2)
  float jacAcc = 0.0f;

  for (int blk = 0; blk < 6; ++blk) {
    __syncthreads();

    // ---- stage weights fp32 -> fp16 into LDS, pre-swizzled into B-operand order ----
    for (int idx = tid; idx < 32 * 64; idx += THREADS) {        // W1: K padded 8->32
      const int k = idx >> 6, col = idx & 63;
      w1s[bswiz(k, col)] = (k < 8) ? (_Float16)W1g[blk * 512 + k * 64 + col]
                                   : (_Float16)0.0f;
    }
    for (int idx = tid; idx < 64 * 64; idx += THREADS) {        // W2
      const int k = idx >> 6, col = idx & 63;
      w2s[bswiz(k, col)] = (_Float16)W2g[blk * 4096 + k * 64 + col];
    }
    for (int idx = tid; idx < 64 * 128; idx += THREADS) {       // W3, cols permuted
      const int k = idx >> 7, cp = idx & 127;                   // cp = j*32 + q
      const int j = cp >> 5, q = cp & 31;
      const int page = (cp >> 4) * 2 + (k >> 5);
      const int dl   = (cp & 15) + (k & 16);
      w3s[page * 512 + dl * 16 + (k & 15)] =
          (q < 31) ? (_Float16)W3g[(blk * 64 + k) * 124 + j * 31 + q]
                   : (_Float16)0.0f;
    }
    if (tid < 64)       b12[tid] = b1g[blk * 64 + tid];
    else if (tid < 128) b12[tid] = b2g[blk * 64 + (tid - 64)];
    if (tid < 128) {                                            // b3 permuted
      const int j = tid >> 5, q = tid & 31;
      b3s[tid] = (q < 31) ? b3g[blk * 124 + j * 31 + q] : 0.0f;
    }
    __syncthreads();

    const int ci0 = COND_TBL[blk][0], ci1 = COND_TBL[blk][1],
              ci2 = COND_TBL[blk][2], ci3 = COND_TBL[blk][3];

    // ================= GEMM1: (16x8) @ (8x64), K padded to 32 =================
    v16h a1v;
    #pragma unroll
    for (int h = 0; h < 16; ++h) a1v[h] = (_Float16)0.0f;
    if (lane < 16) {
      a1v[0] = (_Float16)lds_x[rWG * 8 + ci0];
      a1v[1] = (_Float16)lds_x[rWG * 8 + ci1];
      a1v[2] = (_Float16)lds_x[rWG * 8 + ci2];
      a1v[3] = (_Float16)lds_x[rWG * 8 + ci3];
      a1v[4] = (_Float16)lds_c[rWG * 4 + 0];
      a1v[5] = (_Float16)lds_c[rWG * 4 + 1];
      a1v[6] = (_Float16)lds_c[rWG * 4 + 2];
      a1v[7] = (_Float16)lds_c[rWG * 4 + 3];
    }
    {
      v8f acc[4];
      #pragma unroll
      for (int g = 0; g < 4; ++g) {
        const float bias1 = b12[colL + 16 * g];
        #pragma unroll
        for (int v = 0; v < 8; ++v) acc[g][v] = bias1;
      }
      #pragma unroll
      for (int g = 0; g < 4; ++g) {          // all WMMAs issued before any D use
        const _Float16* bp = w1s + g * 512 + lane * 16;
        const v16h b = cat8(*(const v8h*)bp, *(const v8h*)(bp + 8));
        acc[g] = wmma_f16(a1v, b, acc[g]);
      }
      #pragma unroll
      for (int g = 0; g < 4; ++g) {          // deferred relu + store
        const int col = colL + 16 * g;
        #pragma unroll
        for (int v = 0; v < 8; ++v)
          hA[(wave * 16 + v + mHi) * HSTRIDE + col] = (_Float16)lrelu(acc[g][v]);
      }
    }

    // ================= GEMM2: (16x64) @ (64x64) =================
    {
      const _Float16* ap = hA + rWG * HSTRIDE + 2 * mHi;
      const v16h a0 = cat8(*(const v8h*)(ap),      *(const v8h*)(ap + 8));
      const v16h a1 = cat8(*(const v8h*)(ap + 32), *(const v8h*)(ap + 40));
      v8f acc[4];
      #pragma unroll
      for (int g = 0; g < 4; ++g) {
        const float bias2 = b12[64 + colL + 16 * g];
        #pragma unroll
        for (int v = 0; v < 8; ++v) acc[g][v] = bias2;
      }
      #pragma unroll
      for (int g = 0; g < 4; ++g) {
        const _Float16* b0p = w2s + (g * 2 + 0) * 512 + lane * 16;
        const _Float16* b1p = w2s + (g * 2 + 1) * 512 + lane * 16;
        const v16h b0 = cat8(*(const v8h*)b0p, *(const v8h*)(b0p + 8));
        const v16h b1 = cat8(*(const v8h*)b1p, *(const v8h*)(b1p + 8));
        acc[g] = wmma_f16(a0, b0, acc[g]);
        acc[g] = wmma_f16(a1, b1, acc[g]);
      }
      #pragma unroll
      for (int g = 0; g < 4; ++g) {
        const int col = colL + 16 * g;
        #pragma unroll
        for (int v = 0; v < 8; ++v)
          hB[(wave * 16 + v + mHi) * HSTRIDE + col] = (_Float16)lrelu(acc[g][v]);
      }
    }

    // ================= GEMM3: (16x64) @ (64x128[perm 124]) =================
    {
      const _Float16* ap = hB + rWG * HSTRIDE + 2 * mHi;
      const v16h a0 = cat8(*(const v8h*)(ap),      *(const v8h*)(ap + 8));
      const v16h a1 = cat8(*(const v8h*)(ap + 32), *(const v8h*)(ap + 40));
      v8f acc[8];
      #pragma unroll
      for (int g = 0; g < 8; ++g) {
        const float bias3 = b3s[colL + 16 * g];
        #pragma unroll
        for (int v = 0; v < 8; ++v) acc[g][v] = bias3;
      }
      #pragma unroll
      for (int g = 0; g < 8; ++g) {
        const _Float16* b0p = w3s + (g * 2 + 0) * 512 + lane * 16;
        const _Float16* b1p = w3s + (g * 2 + 1) * 512 + lane * 16;
        const v16h b0 = cat8(*(const v8h*)b0p, *(const v8h*)(b0p + 8));
        const v16h b1 = cat8(*(const v8h*)b1p, *(const v8h*)(b1p + 8));
        acc[g] = wmma_f16(a0, b0, acc[g]);
        acc[g] = wmma_f16(a1, b1, acc[g]);
      }
      #pragma unroll
      for (int g = 0; g < 8; ++g) {
        const int col = colL + 16 * g;
        #pragma unroll
        for (int v = 0; v < 8; ++v)
          outb[(wave * 16 + v + mHi) * OSTRIDE + col] = acc[g][v];
      }
    }

    // ================= Rational-quadratic spline (f32 VALU) =================
    const int db  = (lane >> 4) * 2;
    const int tj0 = TRAFO_TBL[blk][db];
    const int tj1 = TRAFO_TBL[blk][db + 1];
    float outv[2], ladv[2], xinv[2];
    bool myIn = true;
    #pragma unroll
    for (int t = 0; t < 2; ++t) {
      const int j  = db + t;
      const int xd = (t == 0) ? tj0 : tj1;
      const float xin = lds_x[rWG * 8 + xd];
      xinv[t] = xin;
      myIn = myIn && (xin >= 0.0f) && (xin <= 1.0f);
      const float xc = fminf(fmaxf(xin, 0.0f), 1.0f);

      // params: 31 floats at a 16B-aligned base (col-permuted GEMM3 output)
      float pr[32];
      {
        const v4f* pv = (const v4f*)(outb + (wave * 16 + rA) * OSTRIDE + j * 32);
        #pragma unroll
        for (int q = 0; q < 8; ++q) {
          const v4f vq = pv[q];
          pr[q * 4 + 0] = vq[0]; pr[q * 4 + 1] = vq[1];
          pr[q * 4 + 2] = vq[2]; pr[q * 4 + 3] = vq[3];
        }
      }

      float cw[11], ch[11], dv[11];
      { // widths softmax -> cumulative cw
        float ew[10], mx = pr[0], s = 0.0f;
        #pragma unroll
        for (int k = 1; k < 10; ++k) mx = fmaxf(mx, pr[k]);
        #pragma unroll
        for (int k = 0; k < 10; ++k) { ew[k] = __expf(pr[k] - mx); s += ew[k]; }
        const float inv = 1.0f / s;
        cw[0] = 0.0f; float a = 0.0f;
        #pragma unroll
        for (int k = 0; k < 10; ++k) { a += 0.001f + 0.99f * ew[k] * inv; cw[k + 1] = a; }
        cw[10] = 1.0f;
      }
      { // heights softmax -> cumulative ch
        float ew[10], mx = pr[10], s = 0.0f;
        #pragma unroll
        for (int k = 1; k < 10; ++k) mx = fmaxf(mx, pr[10 + k]);
        #pragma unroll
        for (int k = 0; k < 10; ++k) { ew[k] = __expf(pr[10 + k] - mx); s += ew[k]; }
        const float inv = 1.0f / s;
        ch[0] = 0.0f; float a = 0.0f;
        #pragma unroll
        for (int k = 0; k < 10; ++k) { a += 0.001f + 0.99f * ew[k] * inv; ch[k + 1] = a; }
        ch[10] = 1.0f;
      }
      #pragma unroll
      for (int k = 0; k < 11; ++k)
        dv[k] = (0.001f + softplusf(pr[20 + k])) * DINV;

      int bidx = 0;
      #pragma unroll
      for (int k = 1; k < 10; ++k) bidx += (xc >= cw[k]) ? 1 : 0;

      float in_cw = cw[0], in_cw1 = cw[1], in_ch = ch[0], in_ch1 = ch[1];
      float dk = dv[0], dk1 = dv[1];
      #pragma unroll
      for (int k = 1; k < 10; ++k) {
        const bool sel = (bidx == k);
        in_cw  = sel ? cw[k]     : in_cw;
        in_cw1 = sel ? cw[k + 1] : in_cw1;
        in_ch  = sel ? ch[k]     : in_ch;
        in_ch1 = sel ? ch[k + 1] : in_ch1;
        dk     = sel ? dv[k]     : dk;
        dk1    = sel ? dv[k + 1] : dk1;
      }
      const float in_bw    = in_cw1 - in_cw;
      const float in_h     = in_ch1 - in_ch;
      const float in_delta = in_h / in_bw;
      const float theta = (xc - in_cw) / in_bw;
      const float omt   = 1.0f - theta;
      const float tomt  = theta * omt;
      const float num   = in_h * (in_delta * theta * theta + dk * tomt);
      const float den   = in_delta + (dk + dk1 - 2.0f * in_delta) * tomt;
      outv[t] = in_ch + num / den;
      const float dnum = in_delta * in_delta *
                         (dk1 * theta * theta + 2.0f * in_delta * tomt + dk * omt * omt);
      ladv[t] = __logf(dnum / (den * den));   // log(dnum) - 2*log(den)
    }

    const int otherIn = __shfl_xor((int)myIn, 16, 32);
    const bool inside = myIn && (otherIn != 0);
    lds_x[rWG * 8 + tj0] = inside ? outv[0] : xinv[0];
    lds_x[rWG * 8 + tj1] = inside ? outv[1] : xinv[1];
    jacAcc += inside ? (ladv[0] + ladv[1]) : 0.0f;
  }

  const float jtot = jacAcc + __shfl_xor(jacAcc, 16, 32);
  if (lane < 16)
    jac_out[row0 + wave * 16 + rA] = jtot;
}

extern "C" void kernel_launch(void* const* d_in, const int* in_sizes, int n_in,
                              void* d_out, int out_size, void* d_ws, size_t ws_size,
                              hipStream_t stream) {
  const float* xg  = (const float*)d_in[0];
  const float* cg  = (const float*)d_in[1];
  const float* W1g = (const float*)d_in[2];
  const float* b1g = (const float*)d_in[3];
  const float* W2g = (const float*)d_in[4];
  const float* b2g = (const float*)d_in[5];
  const float* W3g = (const float*)d_in[6];
  const float* b3g = (const float*)d_in[7];
  float* jac = (float*)d_out;

  const int n = in_sizes[0] / 8;
  const int grid = (n + ROWS_PER_WG - 1) / ROWS_PER_WG;

  (void)hipFuncSetAttribute((const void*)flow_wmma_kernel,
                            hipFuncAttributeMaxDynamicSharedMemorySize,
                            LDS_BYTES);
  flow_wmma_kernel<<<grid, THREADS, LDS_BYTES, stream>>>(
      xg, cg, W1g, b1g, W2g, b2g, W3g, b3g, jac);
}
